// OptimizedSATAttention_76613626626551
// MI455X (gfx1250) — matrix operand order repaired and checked by reading, not compiled, not run on hardware
//
#include <hip/hip_runtime.h>
#include <hip/hip_bf16.h>

typedef __attribute__((ext_vector_type(16))) __bf16 v16bf;
typedef __attribute__((ext_vector_type(8)))  __bf16 v8bf;
typedef __attribute__((ext_vector_type(4)))  __bf16 v4bf;
typedef __attribute__((ext_vector_type(8)))  float   v8f;

#define D_MODEL 1024
#define NSEQ    2048
#define BATCH   2
#define NHEADS  16
#define DK      64
#define MROWS   (BATCH * NSEQ)      // 4096
#define KSTEPS  (D_MODEL / 32)      // 32
#define W_ELEMS ((size_t)D_MODEL * D_MODEL)

struct ProjArgs {
    const unsigned short* x[5];   // bf16 activations (pre-converted)
    const unsigned short* w[5];   // bf16 weights (pre-converted)
    const float*          b[5];   // f32 biases
};

// ---- helpers -------------------------------------------------------------

__device__ __forceinline__ v16bf ld2_v16(const __bf16* p0, const __bf16* p1) {
    v8bf lo = *(const v8bf*)p0;
    v8bf hi = *(const v8bf*)p1;
    return __builtin_shufflevector(lo, hi, 0,1,2,3,4,5,6,7,8,9,10,11,12,13,14,15);
}

__device__ __forceinline__ v8f wmma_bf16(v16bf a, v16bf b, v8f c) {
    return __builtin_amdgcn_wmma_f32_16x16x32_bf16(false, a, false, b, (short)0, c,
                                                   false, false);
}

// gfx1250 async copy: 16B per lane, global -> LDS, tracked by ASYNCcnt
__device__ __forceinline__ void async_b128(unsigned lds_off, const __bf16* g) {
    unsigned long long ga = (unsigned long long)(size_t)g;
    asm volatile("global_load_async_to_lds_b128 %0, %1, off"
                 :: "v"(lds_off), "v"(ga) : "memory");
}
__device__ __forceinline__ void wait_async0() {
    asm volatile("s_wait_asynccnt 0x0" ::: "memory");
}

// ---- kernel 0: f32 -> bf16 conversion pass (exact-size grids) ------------
__global__ __launch_bounds__(256)
void cvt_bf16_kernel(const float* __restrict__ src, unsigned short* __restrict__ dst) {
    const size_t i = ((size_t)blockIdx.x * 256 + threadIdx.x) * 4;
    float4 v = *(const float4*)(src + i);
    v4bf r;
    r[0] = (__bf16)v.x; r[1] = (__bf16)v.y; r[2] = (__bf16)v.z; r[3] = (__bf16)v.w;
    *(v4bf*)(reinterpret_cast<__bf16*>(dst) + i) = r;
}

// ---- kernel 1: fused Q/K/V/SQ/SK projections (Y = X @ W^T + b, bf16) -----
// grid = (MROWS/128, D_MODEL/64, 5), block = 256 (8 waves)
// All operands bf16; tiles staged by async-to-LDS DMA, double buffered.
__global__ __launch_bounds__(256)
void proj_gemm_kernel(ProjArgs args, unsigned short* ws_raw) {
    __shared__ __bf16 sX[2][128][32];   // 8 KB x2
    __shared__ __bf16 sW[2][64][32];    // 4 KB x2

    const int t    = threadIdx.x;
    const int lane = t & 31;
    const int wave = t >> 5;
    const int half = lane >> 4;
    const int l16  = lane & 15;
    const int m0   = blockIdx.x * 128;
    const int n0   = blockIdx.y * 64;
    const int z    = blockIdx.z;

    const __bf16* X  = reinterpret_cast<const __bf16*>(args.x[z]);
    const __bf16* W  = reinterpret_cast<const __bf16*>(args.w[z]);
    const float*  Bv = args.b[z];
    __bf16* out = reinterpret_cast<__bf16*>(ws_raw) + (size_t)z * MROWS * D_MODEL;

    // staging coordinates (X: two 16B async chunks per thread, W: one)
    const int xr = t >> 1;             // 0..127
    const int xc = (t & 1) * 16;       // 0 / 16
    const int wr = t >> 2;             // 0..63
    const int wc = (t & 3) * 8;        // 0,8,16,24
    const __bf16* xsrc = X + (size_t)(m0 + xr) * D_MODEL + xc;
    const __bf16* wsrc = W + (size_t)(n0 + wr) * D_MODEL + wc;

    auto stage_async = [&](int k, int buf) {
        unsigned lx = (unsigned)(size_t)&sX[buf][xr][xc];
        async_b128(lx,      xsrc + k);
        async_b128(lx + 16, xsrc + k + 8);
        async_b128((unsigned)(size_t)&sW[buf][wr][wc], wsrc + k);
    };

    v8f acc[4] = {};
    stage_async(0, 0);
    wait_async0();
    __syncthreads();

    const int arow = wave * 16 + l16;
    for (int kt = 0; kt < KSTEPS; ++kt) {
        const int buf = kt & 1;
        if (kt + 1 < KSTEPS) stage_async((kt + 1) * 32, buf ^ 1);  // DMA overlaps WMMA
        __builtin_amdgcn_sched_barrier(0);

        v16bf a  = ld2_v16(&sX[buf][arow][16 * half], &sX[buf][arow][16 * half + 8]);
        v16bf b0 = ld2_v16(&sW[buf][ 0 + l16][16 * half], &sW[buf][ 0 + l16][16 * half + 8]);
        v16bf b1 = ld2_v16(&sW[buf][16 + l16][16 * half], &sW[buf][16 + l16][16 * half + 8]);
        v16bf b2 = ld2_v16(&sW[buf][32 + l16][16 * half], &sW[buf][32 + l16][16 * half + 8]);
        v16bf b3 = ld2_v16(&sW[buf][48 + l16][16 * half], &sW[buf][48 + l16][16 * half + 8]);
        acc[0] = wmma_bf16(a, b0, acc[0]);
        acc[1] = wmma_bf16(a, b1, acc[1]);
        acc[2] = wmma_bf16(a, b2, acc[2]);
        acc[3] = wmma_bf16(a, b3, acc[3]);

        wait_async0();
        __syncthreads();
    }

#pragma unroll
    for (int j = 0; j < 4; ++j) {
        const int col  = n0 + 16 * j + l16;
        const float bv = Bv[col];
        const int h = col >> 6;
        const int d = col & 63;
#pragma unroll
        for (int r = 0; r < 8; ++r) {
            const int row = m0 + wave * 16 + r + 8 * half;
            const int bb  = row >> 11;
            const int nn  = row & 2047;
            out[(((size_t)(bb * NHEADS + h)) * NSEQ + nn) * DK + d] =
                (__bf16)(acc[j][r] + bv);
        }
    }
}

// ---- kernel 2: dual-stream flash attention -------------------------------
// grid = (NSEQ/128, BATCH*NHEADS), block = 256 (8 waves, one 16-query tile each)
// K tiles staged by async-to-LDS DMA; V transposed via register path.
__global__ __launch_bounds__(256)
void attn_kernel(const unsigned short* qc_, const unsigned short* kc_,
                 const unsigned short* vc_, const unsigned short* qs_,
                 const unsigned short* ks_, unsigned short* ao_) {
    __shared__ __bf16 sKC[2][32][64];   // 4 KB x2
    __shared__ __bf16 sKS[2][32][64];   // 4 KB x2
    __shared__ __bf16 sVT[2][64][32];   // 4 KB x2 (V transposed [dim][key])
    __shared__ __bf16 sP[8][16][32];    // per-wave P relayout buffer, 8 KB

    const int lane = threadIdx.x & 31;
    const int wave = threadIdx.x >> 5;
    const int half = lane >> 4;
    const int l16  = lane & 15;
    const int bh   = blockIdx.y;                 // b*16 + h
    const size_t base = (size_t)bh * NSEQ * DK;

    const __bf16* qc = reinterpret_cast<const __bf16*>(qc_) + base;
    const __bf16* kc = reinterpret_cast<const __bf16*>(kc_) + base;
    const __bf16* vc = reinterpret_cast<const __bf16*>(vc_) + base;
    const __bf16* qs = reinterpret_cast<const __bf16*>(qs_) + base;
    const __bf16* ks = reinterpret_cast<const __bf16*>(ks_) + base;
    __bf16* ao = reinterpret_cast<__bf16*>(ao_);

    const int q0 = (blockIdx.x * 8 + wave) * 16;

    // Q tiles in A layout, resident for the whole key sweep
    const __bf16* qrc = qc + (size_t)(q0 + l16) * DK;
    const __bf16* qrs = qs + (size_t)(q0 + l16) * DK;
    v16bf qcA[2], qsA[2];
#pragma unroll
    for (int s = 0; s < 2; ++s) {
        qcA[s] = ld2_v16(qrc + 32 * s + 16 * half, qrc + 32 * s + 16 * half + 8);
        qsA[s] = ld2_v16(qrs + 32 * s + 16 * half, qrs + 32 * s + 16 * half + 8);
    }

    float run_max[8], run_sum[8];
    v8f o[4] = {};
#pragma unroll
    for (int r = 0; r < 8; ++r) { run_max[r] = -3.0e38f; run_sum[r] = 0.f; }

    const int skey = threadIdx.x >> 3;        // 0..31
    const int sd0  = (threadIdx.x & 7) * 8;   // 0..56

    auto stage_async_k = [&](int kt, int buf) {
        const size_t off = (size_t)(kt + skey) * DK + sd0;
        async_b128((unsigned)(size_t)&sKC[buf][skey][sd0], kc + off);
        async_b128((unsigned)(size_t)&sKS[buf][skey][sd0], ks + off);
    };
    v8bf rv;
    auto load_v  = [&](int kt) { rv = *(const v8bf*)(vc + (size_t)(kt + skey) * DK + sd0); };
    auto store_v = [&](int buf) {
#pragma unroll
        for (int i = 0; i < 8; ++i) sVT[buf][sd0 + i][skey] = rv[i];
    };

    stage_async_k(0, 0);
    load_v(0);
    wait_async0();
    store_v(0);
    __syncthreads();

    for (int kt = 0; kt < NSEQ; kt += 32) {
        const int buf = (kt >> 5) & 1;
        if (kt + 32 < NSEQ) { stage_async_k(kt + 32, buf ^ 1); load_v(kt + 32); }
        __builtin_amdgcn_sched_barrier(0);

        // scores: S = (Qc Kc^T + Qs Ks^T)/8   (two 16x16 key tiles)
        v8f s0 = {}, s1 = {};
#pragma unroll
        for (int s = 0; s < 2; ++s) {
            const int co = 32 * s + 16 * half;
            v16bf bc0 = ld2_v16(&sKC[buf][l16][co],      &sKC[buf][l16][co + 8]);
            v16bf bs0 = ld2_v16(&sKS[buf][l16][co],      &sKS[buf][l16][co + 8]);
            v16bf bc1 = ld2_v16(&sKC[buf][16 + l16][co], &sKC[buf][16 + l16][co + 8]);
            v16bf bs1 = ld2_v16(&sKS[buf][16 + l16][co], &sKS[buf][16 + l16][co + 8]);
            s0 = wmma_bf16(qcA[s], bc0, s0);
            s0 = wmma_bf16(qsA[s], bs0, s0);
            s1 = wmma_bf16(qcA[s], bc1, s1);
            s1 = wmma_bf16(qsA[s], bs1, s1);
        }

        // online softmax (rows r+8*half, columns across lanes 0..15 of each half)
        float tmax[8];
#pragma unroll
        for (int r = 0; r < 8; ++r) {
            s0[r] *= 0.125f; s1[r] *= 0.125f;
            tmax[r] = fmaxf(s0[r], s1[r]);
        }
#pragma unroll
        for (int m = 1; m < 16; m <<= 1)
#pragma unroll
            for (int r = 0; r < 8; ++r)
                tmax[r] = fmaxf(tmax[r], __shfl_xor(tmax[r], m, 32));

        float p0[8], p1[8], alpha[8], rsum[8];
#pragma unroll
        for (int r = 0; r < 8; ++r) {
            const float nm = fmaxf(run_max[r], tmax[r]);
            alpha[r]   = __expf(run_max[r] - nm);
            run_max[r] = nm;
            p0[r] = __expf(s0[r] - nm);
            p1[r] = __expf(s1[r] - nm);
            rsum[r] = p0[r] + p1[r];
        }
#pragma unroll
        for (int m = 1; m < 16; m <<= 1)
#pragma unroll
            for (int r = 0; r < 8; ++r)
                rsum[r] += __shfl_xor(rsum[r], m, 32);

#pragma unroll
        for (int r = 0; r < 8; ++r) {
            run_sum[r] = run_sum[r] * alpha[r] + rsum[r];
#pragma unroll
            for (int j = 0; j < 4; ++j) o[j][r] *= alpha[r];
            // spill P in C/D layout; reloaded below in A layout (same-wave LDS
            // ops are in-order on CDNA5)
            sP[wave][r + 8 * half][l16]      = (__bf16)p0[r];
            sP[wave][r + 8 * half][16 + l16] = (__bf16)p1[r];
        }

        v16bf pA = ld2_v16(&sP[wave][l16][16 * half], &sP[wave][l16][16 * half + 8]);
#pragma unroll
        for (int j = 0; j < 4; ++j) {
            const int n = 16 * j + l16;
            o[j] = wmma_bf16(pA, ld2_v16(&sVT[buf][n][16 * half],
                                         &sVT[buf][n][16 * half + 8]), o[j]);
        }

        if (kt + 32 < NSEQ) store_v(buf ^ 1);
        wait_async0();
        __syncthreads();
    }

    // finalize: divide by running sum, store bf16 at [b][n][h*64+d]
    const int b = bh >> 4, h = bh & 15;
#pragma unroll
    for (int r = 0; r < 8; ++r) {
        const float inv = 1.0f / run_sum[r];
        const size_t row = (size_t)(b * NSEQ + q0 + r + 8 * half);
        __bf16* orow = ao + row * D_MODEL + h * DK + l16;
#pragma unroll
        for (int j = 0; j < 4; ++j) orow[16 * j] = (__bf16)(o[j][r] * inv);
    }
}

// ---- kernel 3: output projection (f32 out = AO @ Wo^T + bo) --------------
// grid = (MROWS/128, D_MODEL/64), block = 256. Async-staged, double buffered.
__global__ __launch_bounds__(256)
void out_gemm_kernel(const unsigned short* ao_, const unsigned short* wbf_,
                     const float* Bv, float* out) {
    __shared__ __bf16 sA[2][128][32];   // 8 KB x2
    __shared__ __bf16 sW[2][64][32];    // 4 KB x2

    const int t    = threadIdx.x;
    const int lane = t & 31;
    const int wave = t >> 5;
    const int half = lane >> 4;
    const int l16  = lane & 15;
    const int m0   = blockIdx.x * 128;
    const int n0   = blockIdx.y * 64;

    const __bf16* A = reinterpret_cast<const __bf16*>(ao_);
    const __bf16* W = reinterpret_cast<const __bf16*>(wbf_);

    const int xr = t >> 1;
    const int xc = (t & 1) * 16;
    const int wr = t >> 2;
    const int wc = (t & 3) * 8;
    const __bf16* asrc = A + (size_t)(m0 + xr) * D_MODEL + xc;
    const __bf16* wsrc = W + (size_t)(n0 + wr) * D_MODEL + wc;

    auto stage_async = [&](int k, int buf) {
        unsigned la = (unsigned)(size_t)&sA[buf][xr][xc];
        async_b128(la,      asrc + k);
        async_b128(la + 16, asrc + k + 8);
        async_b128((unsigned)(size_t)&sW[buf][wr][wc], wsrc + k);
    };

    v8f acc[4] = {};
    stage_async(0, 0);
    wait_async0();
    __syncthreads();

    const int arow = wave * 16 + l16;
    for (int kt = 0; kt < KSTEPS; ++kt) {
        const int buf = kt & 1;
        if (kt + 1 < KSTEPS) stage_async((kt + 1) * 32, buf ^ 1);
        __builtin_amdgcn_sched_barrier(0);

        v16bf a  = ld2_v16(&sA[buf][arow][16 * half], &sA[buf][arow][16 * half + 8]);
        v16bf b0 = ld2_v16(&sW[buf][ 0 + l16][16 * half], &sW[buf][ 0 + l16][16 * half + 8]);
        v16bf b1 = ld2_v16(&sW[buf][16 + l16][16 * half], &sW[buf][16 + l16][16 * half + 8]);
        v16bf b2 = ld2_v16(&sW[buf][32 + l16][16 * half], &sW[buf][32 + l16][16 * half + 8]);
        v16bf b3 = ld2_v16(&sW[buf][48 + l16][16 * half], &sW[buf][48 + l16][16 * half + 8]);
        acc[0] = wmma_bf16(a, b0, acc[0]);
        acc[1] = wmma_bf16(a, b1, acc[1]);
        acc[2] = wmma_bf16(a, b2, acc[2]);
        acc[3] = wmma_bf16(a, b3, acc[3]);

        wait_async0();
        __syncthreads();
    }

#pragma unroll
    for (int j = 0; j < 4; ++j) {
        const int col  = n0 + 16 * j + l16;
        const float bv = Bv[col];
#pragma unroll
        for (int r = 0; r < 8; ++r) {
            const int row = m0 + wave * 16 + r + 8 * half;
            out[(size_t)row * D_MODEL + col] = acc[j][r] + bv;
        }
    }
}

// ---- launch --------------------------------------------------------------

extern "C" void kernel_launch(void* const* d_in, const int* in_sizes, int n_in,
                              void* d_out, int out_size, void* d_ws, size_t ws_size,
                              hipStream_t stream) {
    const float* x_content = (const float*)d_in[0];
    const float* x_struct  = (const float*)d_in[1];
    const float* Wq  = (const float*)d_in[2];
    const float* bq  = (const float*)d_in[3];
    const float* Wk  = (const float*)d_in[4];
    const float* bk  = (const float*)d_in[5];
    const float* Wv  = (const float*)d_in[6];
    const float* bv  = (const float*)d_in[7];
    const float* Wsq = (const float*)d_in[8];
    const float* bsq = (const float*)d_in[9];
    const float* Wsk = (const float*)d_in[10];
    const float* bsk = (const float*)d_in[11];
    const float* Wo  = (const float*)d_in[12];
    const float* bo  = (const float*)d_in[13];

    unsigned short* ws = (unsigned short*)d_ws;
    const size_t PROJ = (size_t)MROWS * D_MODEL;   // 4M elements

    // ws layout (bf16 elements):
    //   [0..5)*PROJ  : q_c, k_c, v_c, q_s, k_s
    //   5*PROJ       : attention output
    //   6*PROJ       : x_content bf16
    //   7*PROJ       : x_struct  bf16
    //   8*PROJ + i*W_ELEMS : W bf16 (Wq,Wk,Wv,Wsq,Wsk,Wo)
    unsigned short* xc_bf = ws + 6 * PROJ;
    unsigned short* xs_bf = ws + 7 * PROJ;
    unsigned short* w_bf  = ws + 8 * PROJ;

    const float* wsrc[6] = { Wq, Wk, Wv, Wsq, Wsk, Wo };

    // 0) one-time f32 -> bf16 conversion (activations + weights)
    cvt_bf16_kernel<<<dim3(PROJ / 4 / 256), 256, 0, stream>>>(x_content, xc_bf);
    cvt_bf16_kernel<<<dim3(PROJ / 4 / 256), 256, 0, stream>>>(x_struct,  xs_bf);
    for (int i = 0; i < 6; ++i)
        cvt_bf16_kernel<<<dim3(W_ELEMS / 4 / 256), 256, 0, stream>>>(
            wsrc[i], w_bf + (size_t)i * W_ELEMS);

    ProjArgs pa;
    pa.x[0] = xc_bf; pa.x[1] = xc_bf; pa.x[2] = xc_bf;
    pa.x[3] = xs_bf; pa.x[4] = xs_bf;
    for (int i = 0; i < 5; ++i) pa.w[i] = w_bf + (size_t)i * W_ELEMS;
    pa.b[0] = bq; pa.b[1] = bk; pa.b[2] = bv; pa.b[3] = bsq; pa.b[4] = bsk;

    // 1) five projections -> ws[0..5*PROJ) as bf16, head-split layout
    proj_gemm_kernel<<<dim3(MROWS / 128, D_MODEL / 64, 5), 256, 0, stream>>>(pa, ws);

    // 2) dual-stream flash attention -> ws[5*PROJ..6*PROJ) as bf16 [B,N,D]
    attn_kernel<<<dim3(NSEQ / 128, BATCH * NHEADS), 256, 0, stream>>>(
        ws + 0 * PROJ,  // q_c
        ws + 1 * PROJ,  // k_c
        ws + 2 * PROJ,  // v_c
        ws + 3 * PROJ,  // q_s
        ws + 4 * PROJ,  // k_s
        ws + 5 * PROJ); // attention output

    // 3) output projection -> f32 d_out
    out_gemm_kernel<<<dim3(MROWS / 128, D_MODEL / 64), 256, 0, stream>>>(
        ws + 5 * PROJ, w_bf + 5 * W_ELEMS, bo, (float*)d_out);
}
